// GraphHINGE_34248069218766
// MI455X (gfx1250) — matrix operand order repaired and checked by reading, not compiled, not run on hardware
//
#include <hip/hip_runtime.h>
#include <hip/hip_bf16.h>

// ---------------------------------------------------------------------------
// GraphHINGE forward for MI455X (gfx1250, wave32, WMMA f32_16x16x32_f16)
// B=512, L=32, IN=HID=OUT=128, HEADS=3, NMETA=5, T1=T2=0.2
// Weights are pre-swizzled into WMMA-B fragment order so each lane's 16 halfs
// are one contiguous 32-byte load (global_load_b128 x2, coalesced wave-wide).
// ---------------------------------------------------------------------------

typedef __attribute__((ext_vector_type(16))) _Float16 v16h;
typedef __attribute__((ext_vector_type(8)))  float    v8f;

// ---- WMMA helpers (wave32, one wave computes a 16x16 f32 tile, K=32/step) --

__device__ __forceinline__ v8f wmma_f16(v16h a, v16h b, v8f c) {
    // D = A(16x32 f16) * B(32x16 f16) + C(16x16 f32)
    return __builtin_amdgcn_wmma_f32_16x16x32_f16(
        /*neg_a=*/false, a, /*neg_b=*/false, b,
        /*c_mod=*/(short)0, c, /*reuse_a=*/false, /*reuse_b=*/false);
}

// A fragment: 16x32 f16 tile from row-major buffer (row stride ld halfs),
// tile columns [k0, k0+32). Layout used here (contiguous k): lane -> M = lane&15;
// lane<16 holds K {0..15}, lane>=16 holds K {16..31}.
// The 16-half run is contiguous -> compiler emits 2x ds_load_b128.
__device__ __forceinline__ v16h wmma_ld_a(const _Float16* base, int ld, int k0) {
    const int lane = threadIdx.x & 31;
    const int r  = lane & 15;
    const int hw = lane >> 4;
    v16h a;
#pragma unroll
    for (int t = 0; t < 16; ++t) {
        const int k = hw * 16 + t;
        a[t] = base[r * ld + k0 + k];
    }
    return a;
}

// B fragment from pre-swizzled weights: fragment (ks, j) for this lane is 16
// contiguous halfs at ((ks*8 + j)*32 + lane)*16.
__device__ __forceinline__ v16h wmma_ld_bswz(const _Float16* mat, int ks, int j) {
    const int lane = threadIdx.x & 31;
    return *(const v16h*)(mat + (size_t)((((ks << 3) + j) << 5) + lane) * 16);
}

// D tile scatter: f32 16x16 acc -> C/D layout M = v + 8*(lane>>4), N = lane&15.
__device__ __forceinline__ void wmma_st_f16(_Float16* base, int ld, int c0, v8f acc) {
    const int lane = threadIdx.x & 31;
    const int col = c0 + (lane & 15);
    const int hw  = lane >> 4;
#pragma unroll
    for (int v = 0; v < 8; ++v) base[(v + 8 * hw) * ld + col] = (_Float16)acc[v];
}
__device__ __forceinline__ void wmma_st_f32(float* base, int ld, int c0, v8f acc) {
    const int lane = threadIdx.x & 31;
    const int col = c0 + (lane & 15);
    const int hw  = lane >> 4;
#pragma unroll
    for (int v = 0; v < 8; ++v) base[(v + 8 * hw) * ld + col] = acc[v];
}

// ---- prep: fp32 -> f16, swizzled into WMMA-B fragment order ---------------
// swz index = (((ks*8 + j)*32 + lane)*16 + t), source element
//   k = ks*32 + (lane>>4)*16 + t   (K row)
//   c = j*16 + (lane&15)           (N col)

__global__ void prep_weights(const float* __restrict__ Wt, const float* __restrict__ Ws,
                             const float* __restrict__ Wc,
                             _Float16* __restrict__ WtF, _Float16* __restrict__ WsTF,
                             _Float16* __restrict__ WcF) {
    const int idx = blockIdx.x * 256 + threadIdx.x;   // 5*3*128*128 = 245760
    if (idx >= 245760) return;
    const int t    = idx & 15;
    const int lane = (idx >> 4) & 31;
    const int j    = (idx >> 9) & 7;
    const int ks   = (idx >> 12) & 3;
    const int mh   = idx >> 14;
    const int k = ks * 32 + (lane >> 4) * 16 + t;
    const int c = j * 16 + (lane & 15);
    const size_t base = (size_t)mh * 16384;
    WtF[idx]  = (_Float16)Wt[base + k * 128 + c];      // B[k][c] = Wt[e=k][o=c]
    WcF[idx]  = (_Float16)Wc[base + k * 128 + c];      // B[k][c] = Wc[e=k][o=c]
    WsTF[idx] = (_Float16)Ws[base + c * 128 + k];      // B[k=o][c=e] = Ws[e][o]
}

__global__ void prep_pw1(const float* __restrict__ pw1, _Float16* __restrict__ pw1F) {
    const int idx = blockIdx.x * 256 + threadIdx.x;   // 128*128 = 16384
    if (idx >= 16384) return;
    const int t    = idx & 15;
    const int lane = (idx >> 4) & 31;
    const int j    = (idx >> 9) & 7;
    const int ks   = (idx >> 12) & 3;
    const int k = ks * 32 + (lane >> 4) * 16 + t;
    const int c = j * 16 + (lane & 15);
    pw1F[idx] = (_Float16)pw1[k * 128 + c];
}

// ---- fused gather + correlation + node attention ---------------------------
// One wave per 16-row tile of n = B*L = 16384 node pairs. N = path length.

template <int N>
__global__ __launch_bounds__(32)
void metapath_kernel(const float* __restrict__ emb_user, const float* __restrict__ emb_item,
                     const float* __restrict__ emb_a1,   const float* __restrict__ emb_a2,
                     const float* __restrict__ emb_a3,
                     const int* __restrict__ UI,    const int* __restrict__ IU,
                     const int* __restrict__ UIUI,  const int* __restrict__ IUIU,
                     const int* __restrict__ UIAI1, const int* __restrict__ IAIU1,
                     const int* __restrict__ UIAI2, const int* __restrict__ IAIU2,
                     const int* __restrict__ UIAI3, const int* __restrict__ IAIU3,
                     const _Float16* __restrict__ WtF, const _Float16* __restrict__ WsTF,
                     const _Float16* __restrict__ WcF,
                     float* __restrict__ zbuf) {
    constexpr int P = 2 * N - 1;                    // 3 or 7
    __shared__ _Float16 Hs[16 * 7 * 128];           // corr features, f16 (28 KB)
    __shared__ _Float16 hiS[16 * 128];              // hi tile, f16
    __shared__ float    qS[16 * 128];               // q tile, f32
    __shared__ float    alphaS[16 * 8];             // softmax weights
    __shared__ _Float16 ctxS[16 * 128];             // attended features, f16

    const int lane = threadIdx.x;
    const int m  = (N == 2) ? 0 : (1 + blockIdx.y);
    const int r0 = blockIdx.x * 16;
    const int e0 = lane * 4;

    const int* sidx;
    const int* tidx;
    const float* stab[4];
    const float* ttab[4];
    if (N == 2) {
        sidx = UI;  tidx = IU;
        stab[0] = emb_user; stab[1] = emb_item; stab[2] = emb_user; stab[3] = emb_user;
        ttab[0] = emb_item; ttab[1] = emb_user; ttab[2] = emb_user; ttab[3] = emb_user;
    } else {
        sidx = (m == 1) ? UIUI : (m == 2) ? UIAI1 : (m == 3) ? UIAI2 : UIAI3;
        tidx = (m == 1) ? IUIU : (m == 2) ? IAIU1 : (m == 3) ? IAIU2 : IAIU3;
        const float* atab = (m == 1) ? emb_user : (m == 2) ? emb_a1
                          : (m == 3) ? emb_a2   : emb_a3;
        stab[0] = emb_user; stab[1] = emb_item; stab[2] = atab;     stab[3] = emb_item;
        ttab[0] = emb_item; ttab[1] = atab;     ttab[2] = emb_item; ttab[3] = emb_user;
    }

    // ---- gather embeddings + cross-correlation -> Hs (f16) ----
    for (int i = 0; i < 16; ++i) {
        const int n = r0 + i;
        float s[N][4], t[N][4];
#pragma unroll
        for (int j = 0; j < N; ++j) {
            {
                const int row = sidx[n * N + j];
                const float4 v = *(const float4*)(stab[j] + (size_t)row * 128 + e0);
                s[j][0] = v.x; s[j][1] = v.y; s[j][2] = v.z; s[j][3] = v.w;
            }
            {
                const int row = tidx[n * N + j];
                const float4 v = *(const float4*)(ttab[j] + (size_t)row * 128 + e0);
                t[j][0] = v.x; t[j][1] = v.y; t[j][2] = v.z; t[j][3] = v.w;
            }
        }
#pragma unroll
        for (int p = 0; p < P; ++p) {
#pragma unroll
            for (int c = 0; c < 4; ++c) {
                float acc = 0.f;
#pragma unroll
                for (int ii = 0; ii < N; ++ii) {
                    const int js = p + ii - (N - 1);
                    if (js >= 0 && js < N) acc += s[js][c] * t[ii][c];
                }
                Hs[(i * 7 + p) * 128 + e0 + c] = (_Float16)acc;
            }
        }
    }
    __syncthreads();

    // ---- 3-head node attention, WMMA GEMMs, Z accumulated in registers ----
    v8f zacc[8];
#pragma unroll
    for (int j = 0; j < 8; ++j) { v8f zz = {}; zacc[j] = zz; }

    for (int h = 0; h < 3; ++h) {
        const _Float16* Wt_h  = WtF  + (size_t)(m * 3 + h) * 16384;
        const _Float16* WsT_h = WsTF + (size_t)(m * 3 + h) * 16384;
        const _Float16* Wc_h  = WcF  + (size_t)(m * 3 + h) * 16384;

        // hi = H0(16x128) @ Wt (128x128)
#pragma unroll
        for (int j = 0; j < 8; ++j) {
            v8f acc = {};
#pragma unroll
            for (int ks = 0; ks < 4; ++ks) {
                const v16h a = wmma_ld_a(Hs, 7 * 128, ks * 32);   // p = 0 rows
                const v16h b = wmma_ld_bswz(Wt_h, ks, j);
                acc = wmma_f16(a, b, acc);
            }
            wmma_st_f16(hiS, 128, j * 16, acc);
        }
        __syncthreads();

        // q = hi @ Ws^T
#pragma unroll
        for (int j = 0; j < 8; ++j) {
            v8f acc = {};
#pragma unroll
            for (int ks = 0; ks < 4; ++ks) {
                const v16h a = wmma_ld_a(hiS, 128, ks * 32);
                const v16h b = wmma_ld_bswz(WsT_h, ks, j);
                acc = wmma_f16(a, b, acc);
            }
            wmma_st_f32(qS, 128, j * 16, acc);
        }
        __syncthreads();

        // logits[p] = H[p] . q / T1 ; softmax over p   (lanes 0..15: one row each)
        if (lane < 16) {
            float lg[P];
            float mx = -1e30f;
#pragma unroll
            for (int p = 0; p < P; ++p) {
                float acc = 0.f;
                for (int e = 0; e < 128; ++e)
                    acc += (float)Hs[(lane * 7 + p) * 128 + e] * qS[lane * 128 + e];
                acc *= 5.0f;                        // 1/T1
                lg[p] = acc;
                mx = fmaxf(mx, acc);
            }
            float sum = 0.f;
#pragma unroll
            for (int p = 0; p < P; ++p) { lg[p] = __expf(lg[p] - mx); sum += lg[p]; }
            const float inv = 1.0f / sum;
#pragma unroll
            for (int p = 0; p < P; ++p) alphaS[lane * 8 + p] = lg[p] * inv;
        }
        __syncthreads();

        // ctx = sum_p alpha[p] * H[p]
#pragma unroll
        for (int c = 0; c < 4; ++c) {
            const int e = e0 + c;
            for (int i = 0; i < 16; ++i) {
                float acc = 0.f;
#pragma unroll
                for (int p = 0; p < P; ++p)
                    acc += alphaS[i * 8 + p] * (float)Hs[(i * 7 + p) * 128 + e];
                ctxS[i * 128 + e] = (_Float16)acc;
            }
        }
        __syncthreads();

        // Z += ctx @ Wc
#pragma unroll
        for (int j = 0; j < 8; ++j) {
#pragma unroll
            for (int ks = 0; ks < 4; ++ks) {
                const v16h a = wmma_ld_a(ctxS, 128, ks * 32);
                const v16h b = wmma_ld_bswz(Wc_h, ks, j);
                zacc[j] = wmma_f16(a, b, zacc[j]);
            }
        }
        __syncthreads();
    }

    // ---- store Z/HEADS into z (B, 160, 128), row = m*32 + l ----
    {
        const int colb = lane & 15;
        const int hw   = lane >> 4;
#pragma unroll
        for (int j = 0; j < 8; ++j) {
#pragma unroll
            for (int v = 0; v < 8; ++v) {
                const int i = v + 8 * hw;
                const int n = r0 + i;
                const int b = n >> 5, l = n & 31;
                zbuf[((size_t)b * 160 + m * 32 + l) * 128 + j * 16 + colb] =
                    zacc[j][v] * (1.0f / 3.0f);
            }
        }
    }
}

// ---- path attention + final MLP: one wave per batch row --------------------

__global__ __launch_bounds__(32)
void head_kernel(const float* __restrict__ zbuf,
                 const _Float16* __restrict__ pw1F,
                 const float* __restrict__ pb1, const float* __restrict__ pw2,
                 const float* __restrict__ fw1, const float* __restrict__ fb1,
                 const float* __restrict__ fw2, const float* __restrict__ fb2,
                 const float* __restrict__ emb_user, const float* __restrict__ emb_item,
                 const int* __restrict__ UI, const int* __restrict__ IU,
                 float* __restrict__ out) {
    __shared__ _Float16 zS[160 * 128];   // 40 KB, reused by GEMM + weighted sum
    __shared__ float    tS[16 * 128];
    __shared__ float    wS[160];
    __shared__ float    betaS[160];
    __shared__ float    xS[384];
    __shared__ float    yS[128];

    const int b = blockIdx.x;
    const int lane = threadIdx.x;
    const int e0 = lane * 4;

    for (int r = 0; r < 160; ++r) {
        const float4 v = *(const float4*)(zbuf + ((size_t)b * 160 + r) * 128 + e0);
        zS[r * 128 + e0 + 0] = (_Float16)v.x;
        zS[r * 128 + e0 + 1] = (_Float16)v.y;
        zS[r * 128 + e0 + 2] = (_Float16)v.z;
        zS[r * 128 + e0 + 3] = (_Float16)v.w;
    }
    __syncthreads();

    // w = tanh(z @ pw1 + pb1) @ pw2 / T2, 10 tiles of 16 rows
    for (int tile = 0; tile < 10; ++tile) {
        const _Float16* ztile = zS + tile * 16 * 128;
#pragma unroll
        for (int j = 0; j < 8; ++j) {
            v8f acc = {};
#pragma unroll
            for (int ks = 0; ks < 4; ++ks) {
                const v16h a = wmma_ld_a(ztile, 128, ks * 32);
                const v16h bb = wmma_ld_bswz(pw1F, ks, j);
                acc = wmma_f16(a, bb, acc);
            }
            const int col = j * 16 + (lane & 15);
            const int hw  = lane >> 4;
#pragma unroll
            for (int v = 0; v < 8; ++v)
                tS[(v + 8 * hw) * 128 + col] = tanhf(acc[v] + pb1[col]);
        }
        __syncthreads();
        if (lane < 16) {
            float acc = 0.f;
            for (int e = 0; e < 128; ++e) acc += tS[lane * 128 + e] * pw2[e];
            wS[tile * 16 + lane] = acc * 5.0f;      // 1/T2
        }
        __syncthreads();
    }

    // softmax over 160 rows (wave32 reductions)
    float mx = -1e30f;
    for (int r = lane; r < 160; r += 32) mx = fmaxf(mx, wS[r]);
#pragma unroll
    for (int off = 16; off > 0; off >>= 1) mx = fmaxf(mx, __shfl_xor(mx, off, 32));
    float sum = 0.f;
    for (int r = lane; r < 160; r += 32) sum += __expf(wS[r] - mx);
#pragma unroll
    for (int off = 16; off > 0; off >>= 1) sum += __shfl_xor(sum, off, 32);
    const float inv = 1.0f / sum;
    for (int r = lane; r < 160; r += 32) betaS[r] = __expf(wS[r] - mx) * inv;
    __syncthreads();

    // path_out = sum_r beta[r] * z[r]
    float po[4] = {0.f, 0.f, 0.f, 0.f};
    for (int r = 0; r < 160; ++r) {
        const float bb = betaS[r];
#pragma unroll
        for (int c = 0; c < 4; ++c) po[c] += bb * (float)zS[r * 128 + e0 + c];
    }

    // x = [src, tgt, path_out]
    const int su = UI[((size_t)b * 32) * 2];
    const int ti = IU[((size_t)b * 32) * 2];
#pragma unroll
    for (int c = 0; c < 4; ++c) {
        xS[e0 + c]       = emb_user[(size_t)su * 128 + e0 + c];
        xS[128 + e0 + c] = emb_item[(size_t)ti * 128 + e0 + c];
        xS[256 + e0 + c] = po[c];
    }
    __syncthreads();

    // y = relu(x @ fw1 + fb1)  (fw1 rows read as float4, coalesced)
    float yacc[4];
#pragma unroll
    for (int c = 0; c < 4; ++c) yacc[c] = fb1[e0 + c];
    for (int k = 0; k < 384; ++k) {
        const float xk = xS[k];
        const float4 w4 = *(const float4*)(fw1 + (size_t)k * 128 + e0);
        yacc[0] += xk * w4.x; yacc[1] += xk * w4.y;
        yacc[2] += xk * w4.z; yacc[3] += xk * w4.w;
    }
#pragma unroll
    for (int c = 0; c < 4; ++c) yS[e0 + c] = fmaxf(yacc[c], 0.f);
    __syncthreads();

    // out = sigmoid(y . fw2 + fb2)
    float acc = 0.f;
    for (int o = lane; o < 128; o += 32) acc += yS[o] * fw2[o];
#pragma unroll
    for (int off = 16; off > 0; off >>= 1) acc += __shfl_xor(acc, off, 32);
    if (lane == 0) out[b] = 1.0f / (1.0f + __expf(-(acc + fb2[0])));
}

// ---------------------------------------------------------------------------

extern "C" void kernel_launch(void* const* d_in, const int* in_sizes, int n_in,
                              void* d_out, int out_size, void* d_ws, size_t ws_size,
                              hipStream_t stream) {
    const int* UI    = (const int*)d_in[0];
    const int* IU    = (const int*)d_in[1];
    const int* UIUI  = (const int*)d_in[2];
    const int* IUIU  = (const int*)d_in[3];
    const int* UIAI1 = (const int*)d_in[4];
    const int* IAIU1 = (const int*)d_in[5];
    const int* UIAI2 = (const int*)d_in[6];
    const int* IAIU2 = (const int*)d_in[7];
    const int* UIAI3 = (const int*)d_in[8];
    const int* IAIU3 = (const int*)d_in[9];
    const float* emb_user = (const float*)d_in[10];
    const float* emb_item = (const float*)d_in[11];
    const float* emb_a1   = (const float*)d_in[12];
    const float* emb_a2   = (const float*)d_in[13];
    const float* emb_a3   = (const float*)d_in[14];
    const float* Wt  = (const float*)d_in[15];
    const float* Ws  = (const float*)d_in[16];
    const float* Wc  = (const float*)d_in[17];
    const float* pw1 = (const float*)d_in[18];
    const float* pb1 = (const float*)d_in[19];
    const float* pw2 = (const float*)d_in[20];
    const float* fw1 = (const float*)d_in[21];
    const float* fb1 = (const float*)d_in[22];
    const float* fw2 = (const float*)d_in[23];
    const float* fb2 = (const float*)d_in[24];

    // workspace layout (all offsets 32B-aligned for v16h loads)
    char* ws = (char*)d_ws;
    _Float16* WtF  = (_Float16*)(ws);                 // 245760 * 2 = 491520 B
    _Float16* WsTF = (_Float16*)(ws + 491520);
    _Float16* WcF  = (_Float16*)(ws + 983040);
    _Float16* pw1F = (_Float16*)(ws + 1474560);       // 32768 B
    float*    zbuf = (float*)(ws + 1507328);          // 512*160*128*4 = 40 MiB

    prep_weights<<<(245760 + 255) / 256, 256, 0, stream>>>(Wt, Ws, Wc, WtF, WsTF, WcF);
    prep_pw1<<<64, 256, 0, stream>>>(pw1, pw1F);

    dim3 g2(1024, 1);   // metapath 0 (N=2), 16384 rows / 16 per wave
    metapath_kernel<2><<<g2, 32, 0, stream>>>(
        emb_user, emb_item, emb_a1, emb_a2, emb_a3,
        UI, IU, UIUI, IUIU, UIAI1, IAIU1, UIAI2, IAIU2, UIAI3, IAIU3,
        WtF, WsTF, WcF, zbuf);

    dim3 g4(1024, 4);   // metapaths 1..4 (N=4)
    metapath_kernel<4><<<g4, 32, 0, stream>>>(
        emb_user, emb_item, emb_a1, emb_a2, emb_a3,
        UI, IU, UIUI, IUIU, UIAI1, IAIU1, UIAI2, IAIU2, UIAI3, IAIU3,
        WtF, WsTF, WcF, zbuf);

    head_kernel<<<512, 32, 0, stream>>>(
        zbuf, pw1F, pb1, pw2, fw1, fb1, fw2, fb2,
        emb_user, emb_item, UI, IU, (float*)d_out);
}
